// KumaIndependentLatentModel_14508399526053
// MI455X (gfx1250) — compile-verified, matrix operand order checked by reading, not run on hardware
//
#include <hip/hip_runtime.h>
#include <hip/hip_bf16.h>

// ---------------------------------------------------------------------------
// Shapes (fixed by the reference)
// ---------------------------------------------------------------------------
#define BB   64
#define TT   1024
#define EE   300
#define EP   320          // E padded to multiple of 32 (K for input GEMM)
#define HH   200
#define KPH  224          // H padded to multiple of 32 (K for recurrent GEMM)
#define NG   800          // 4*H gate width (N for both GEMMs), multiple of 16
#define MROWS (BB*TT)     // 65536

// ---------------------------------------------------------------------------
// CDNA5 feature detection (device pass only; host pass takes fallbacks)
// ---------------------------------------------------------------------------
#if defined(__HIP_DEVICE_COMPILE__)
#  if __has_builtin(__builtin_amdgcn_global_load_async_to_lds_b128)
#    define HAVE_ASYNC_LDS 1
#  endif
#  if __has_builtin(__builtin_amdgcn_s_wait_asynccnt)
#    define HAVE_WAIT_ASYNC 1
#  endif
#  if __has_builtin(__builtin_amdgcn_tensor_load_to_lds) && \
      __has_builtin(__builtin_amdgcn_s_wait_tensorcnt)
#    define HAVE_TDM 1
#  endif
#endif

// ---------------------------------------------------------------------------
// WMMA / builtin vector types (CDNA5 / gfx1250, wave32)
// ---------------------------------------------------------------------------
typedef __attribute__((ext_vector_type(16))) __bf16       v16bf;
typedef __attribute__((ext_vector_type(8)))  float        v8f;
typedef __attribute__((ext_vector_type(8)))  unsigned int v8u;
typedef __attribute__((ext_vector_type(4)))  unsigned int v4u;
typedef __attribute__((ext_vector_type(8)))  int          v8i;
typedef __attribute__((ext_vector_type(4)))  int          v4i;

typedef __attribute__((address_space(3))) void lds_void;
typedef __attribute__((address_space(1))) void glb_void;
typedef __attribute__((address_space(1))) v4i  glb_v4i;   // async src (global)
typedef __attribute__((address_space(3))) v4i  lds_v4i;   // async dst (LDS)

__device__ __forceinline__ unsigned short f2bf(float f) {
  unsigned int u = __builtin_bit_cast(unsigned int, f);
  u += 0x7FFFu + ((u >> 16) & 1u);          // round-to-nearest-even
  return (unsigned short)(u >> 16);
}

__device__ __forceinline__ float sigmoidf_(float x) {
  return 1.0f / (1.0f + __expf(-x));
}
__device__ __forceinline__ float softplusf_(float x) {
  return (x > 30.f) ? x : log1pf(expf(x));
}

// 16-bit A/B operand loader (16x32 tile, MxK or NxK with K contiguous).
// lane<16 holds row (lane), K in {0..7} U {16..23}; lane>=16 same row, K+8.
// Each dword = two consecutive bf16 along K.  ldk must be even, kbase even.
__device__ __forceinline__ v16bf load_tile_global(const unsigned int* p,
                                                  int row0, int kbase,
                                                  int ldk, int lane) {
  int r = lane & 15;
  int khalf = (lane >> 4) & 1;
  int rowoff = (row0 + r) * ldk;
  v8u u;
#pragma unroll
  for (int j = 0; j < 4; ++j) {
    int k0 = kbase + khalf * 8 + 2 * j;
    int k1 = kbase + 16 + khalf * 8 + 2 * j;
    u[j]     = p[(rowoff + k0) >> 1];
    u[j + 4] = p[(rowoff + k1) >> 1];
  }
  return __builtin_bit_cast(v16bf, u);
}

__device__ __forceinline__ v16bf load_tile_lds(const unsigned int* p,
                                               int row0, int kbase,
                                               int ldk, int lane) {
  int r = lane & 15;
  int khalf = (lane >> 4) & 1;
  int rowoff = (row0 + r) * ldk;
  v8u u;
#pragma unroll
  for (int j = 0; j < 4; ++j) {
    int k0 = kbase + khalf * 8 + 2 * j;
    int k1 = kbase + 16 + khalf * 8 + 2 * j;
    u[j]     = p[(rowoff + k0) >> 1];
    u[j + 4] = p[(rowoff + k1) >> 1];
  }
  return __builtin_bit_cast(v16bf, u);
}

// ---------------------------------------------------------------------------
// Kernel 1: embedding gather -> zero-padded bf16 (MROWS, EP)
// ---------------------------------------------------------------------------
__global__ void pad_embed_kernel(const int* __restrict__ x,
                                 const float* __restrict__ embed,
                                 unsigned short* __restrict__ emb_pad) {
  long long idx = (long long)blockIdx.x * blockDim.x + threadIdx.x;
  long long total = (long long)MROWS * EP;
  if (idx >= total) return;
  int row = (int)(idx / EP);
  int k   = (int)(idx % EP);
  float v = 0.f;
  if (k < EE) {
    int tok = x[row];
    v = embed[(long long)tok * EE + k];
  }
  emb_pad[idx] = f2bf(v);
}

// Kernel 1b: cast/zero-pad a (N,K) f32 weight to (N,Kpad) bf16
__global__ void pad_weight_kernel(const float* __restrict__ src,
                                  unsigned short* __restrict__ dst,
                                  int nrows, int K, int Kpad) {
  long long idx = (long long)blockIdx.x * blockDim.x + threadIdx.x;
  long long total = (long long)nrows * Kpad;
  if (idx >= total) return;
  int row = (int)(idx / Kpad);
  int k   = (int)(idx % Kpad);
  float v = (k < K) ? src[(long long)row * K + k] : 0.f;
  dst[idx] = f2bf(v);
}

// ---------------------------------------------------------------------------
// Kernel 2: input projection GEMM  gin = emb_pad(bf16) @ Wih_pad(bf16)^T + b
//   grid = (ceil(50/8), 4096): block shares one 16-row A strip (async->LDS),
//   each of 8 waves owns one 16x16 output tile column group.
// ---------------------------------------------------------------------------
__global__ __launch_bounds__(256)
void input_proj_kernel(const unsigned short* __restrict__ A,
                       const unsigned short* __restrict__ W,
                       const float* __restrict__ bias,
                       float* __restrict__ out) {
  __shared__ unsigned short a_sh[16 * EP];           // 10,240 B strip
  const int ntiles = NG / 16;                        // 50
  int wave  = threadIdx.x >> 5;
  int lane  = threadIdx.x & 31;
  int mtile = blockIdx.y;
  int ntile = blockIdx.x * 8 + wave;

  // ---- stage A strip (16 rows x EP bf16) into LDS ----
  {
    const v4u* srcv = (const v4u*)(A + (size_t)mtile * 16 * EP);
    v4u* dstv = (v4u*)a_sh;
    const int nchunks = 16 * EP * 2 / 16;            // 640 x 16B
    for (int i = threadIdx.x; i < nchunks; i += blockDim.x) {
#if defined(HAVE_ASYNC_LDS)
      __builtin_amdgcn_global_load_async_to_lds_b128(
          (glb_v4i*)(srcv + i), (lds_v4i*)(dstv + i), 0, 0);
#else
      dstv[i] = srcv[i];
#endif
    }
#if defined(HAVE_ASYNC_LDS)
#  if defined(HAVE_WAIT_ASYNC)
    __builtin_amdgcn_s_wait_asynccnt(0);
#  else
    asm volatile("s_wait_asynccnt 0x0" ::: "memory");
#  endif
#endif
  }
  __syncthreads();

  if (ntile < ntiles) {                              // wave-uniform predicate
    const unsigned int* Au = (const unsigned int*)a_sh;
    const unsigned int* Wu = (const unsigned int*)W;
    v8f acc = {};
#pragma unroll
    for (int kb = 0; kb < EP; kb += 32) {
      if (kb + 64 < EP)   // prefetch the B stream two K-tiles ahead
        __builtin_prefetch(&W[(ntile * 16 + (lane & 15)) * EP + kb + 64], 0, 0);
      v16bf a = load_tile_lds(Au, 0, kb, EP, lane);
      v16bf b = load_tile_global(Wu, ntile * 16, kb, EP, lane);
      acc = __builtin_amdgcn_wmma_f32_16x16x32_bf16(false, a, false, b,
                                                    (short)0, acc, false, false);
    }
    int n  = ntile * 16 + (lane & 15);
    int mb = mtile * 16 + ((lane >> 4) << 3);
    float bv = bias[n];
#pragma unroll
    for (int r = 0; r < 8; ++r) {
      out[(long long)(mb + r) * NG + n] = acc[r] + bv;
    }
  }
}

// ---------------------------------------------------------------------------
// Kernel 3: sequential (bi)LSTM.  grid.x = 2 (dir), block = 512 (16 waves).
//   h (bf16, K-padded), c (f32), gates (f32) live in LDS (~285 KB).
//   Per step: TDM pulls the 64x800 f32 input-projection slab into g_sh,
//   WMMA recurrent GEMM accumulates on top, gate math, masked carry update.
// ---------------------------------------------------------------------------
__global__ __launch_bounds__(512)
void lstm_kernel(const unsigned short* __restrict__ whh_f,
                 const unsigned short* __restrict__ whh_b,
                 const float* __restrict__ gin_f,
                 const float* __restrict__ gin_b,
                 const unsigned char* __restrict__ mask,
                 float* __restrict__ hf,
                 float* __restrict__ hb) {
  __shared__ unsigned int h_sh[BB * KPH / 2];   // bf16 pairs, 28,672 B
  __shared__ float        c_sh[BB * HH];        // 51,200 B
  __shared__ float        g_sh[BB * NG];        // 204,800 B

  int dir = blockIdx.x;
  const unsigned short* W = dir ? whh_b : whh_f;
  const float* gin        = dir ? gin_b : gin_f;
  float* hout             = dir ? hb : hf;

  int tid = threadIdx.x;
  for (int i = tid; i < BB * KPH / 2; i += blockDim.x) h_sh[i] = 0u;
  for (int i = tid; i < BB * HH;      i += blockDim.x) c_sh[i] = 0.f;
  __syncthreads();

  int wave   = tid >> 5;
  int lane   = tid & 31;
  int nwaves = blockDim.x >> 5;
  const unsigned int* Wu = (const unsigned int*)W;
  const int ntiles = NG / 16;   // 50
  const int mtiles = BB / 16;   // 4
  unsigned short* h_sh16 = (unsigned short*)h_sh;

  for (int step = 0; step < TT; ++step) {
    int t = dir ? (TT - 1 - step) : step;

#if defined(HAVE_TDM)
    // ---- TDM: DMA the (64 x 800) f32 gin slab for step t into g_sh ----
    // 2D tile: tile_dim0 = 800 elems (4B), tile_dim1 = 64 rows,
    // row stride = TT*NG elems.  D# per CDNA5 ISA.
    if (wave == 0) {
      unsigned lds_off = (unsigned)(unsigned long long)(lds_void*)g_sh;
      unsigned long long ga =
          (unsigned long long)(const glb_void*)(gin + (size_t)t * NG);
      v4u g0;
      g0[0] = 1u;                                   // count=1, user mode
      g0[1] = lds_off;                              // lds_addr
      g0[2] = (unsigned)(ga & 0xFFFFFFFFu);         // global_addr[31:0]
      g0[3] = (unsigned)((ga >> 32) & 0x1FFFFFFu)   // global_addr[56:32]
              | (2u << 30);                         // type=2 ("image")
      v8i g1;
      g1[0] = (int)(2u << 16);                      // data_size=4B
      g1[1] = (int)((unsigned)NG << 16);            // tensor_dim0 lo16
      g1[2] = (int)((unsigned)BB << 16);            // dim0 hi=0 | tensor_dim1 lo16
      g1[3] = (int)((unsigned)NG << 16);            // dim1 hi=0 | tile_dim0
      g1[4] = BB;                                   // tile_dim1=64, tile_dim2=0
      g1[5] = TT * NG;                              // tensor_dim0_stride lo32
      g1[6] = 0;                                    // stride0 hi | stride1 lo
      g1[7] = 0;
      v4i g2 = {};
      v4i g3 = {};
#  if __clang_major__ >= 23
      v8i g4 = {};
      __builtin_amdgcn_tensor_load_to_lds(g0, g1, g2, g3, g4, 0);
#  else
      __builtin_amdgcn_tensor_load_to_lds(g0, g1, g2, g3, 0);
#  endif
      __builtin_amdgcn_s_wait_tensorcnt(0);
    }
    __syncthreads();
#endif

    // ---- recurrent GEMM: gates (+)= h @ Whh^T ----
    for (int tile = wave; tile < mtiles * ntiles; tile += nwaves) {
      int mtile = tile / ntiles;
      int ntile = tile % ntiles;
      v8f acc = {};
#pragma unroll
      for (int kb = 0; kb < KPH; kb += 32) {
        v16bf a = load_tile_lds(h_sh, mtile * 16, kb, KPH, lane);
        v16bf b = load_tile_global(Wu, ntile * 16, kb, KPH, lane);
        acc = __builtin_amdgcn_wmma_f32_16x16x32_bf16(false, a, false, b,
                                                      (short)0, acc,
                                                      false, false);
      }
      int n  = ntile * 16 + (lane & 15);
      int mb = mtile * 16 + ((lane >> 4) << 3);
#pragma unroll
      for (int r = 0; r < 8; ++r) {
        int bm = mb + r;   // batch index
#if defined(HAVE_TDM)
        g_sh[bm * NG + n] += acc[r];
#else
        g_sh[bm * NG + n] = acc[r] + gin[((long long)bm * TT + t) * NG + n];
#endif
      }
    }
    __syncthreads();

    // ---- elementwise gate math + masked carry update ----
    for (int u = tid; u < BB * HH; u += blockDim.x) {
      int b = u / HH;
      int j = u % HH;
      unsigned char mt = mask[b * TT + t];
      float gi = g_sh[b * NG + j];
      float gf = g_sh[b * NG + HH + j];
      float gg = g_sh[b * NG + 2 * HH + j];
      float go = g_sh[b * NG + 3 * HH + j];
      float c_old = c_sh[u];
      float c_new = sigmoidf_(gf) * c_old + sigmoidf_(gi) * tanhf(gg);
      float h_new = sigmoidf_(go) * tanhf(c_new);
      if (mt) {
        c_sh[u] = c_new;
        h_sh16[b * KPH + j] = f2bf(h_new);
      }
      hout[((long long)b * TT + t) * HH + j] = mt ? h_new : 0.f;
    }
    __syncthreads();
  }
}

// ---------------------------------------------------------------------------
// Kernel 4: Kumaraswamy head -> z_probs (B,T)
// ---------------------------------------------------------------------------
__global__ void head_kernel(const float* __restrict__ hf,
                            const float* __restrict__ hb,
                            const float* __restrict__ Wa,
                            const float* __restrict__ ba,
                            const float* __restrict__ Wb,
                            const float* __restrict__ bb,
                            const unsigned char* __restrict__ mask,
                            float* __restrict__ zp) {
  long long idx = (long long)blockIdx.x * blockDim.x + threadIdx.x;
  if (idx >= (long long)MROWS) return;
  const float* hfr = hf + idx * HH;
  const float* hbr = hb + idx * HH;
  float sa = 0.f, sb = 0.f;
  for (int j = 0; j < HH; ++j) {
    float v = hfr[j] * 100.f;          // hs = h / 0.01
    sa += v * Wa[j];
    sb += v * Wb[j];
  }
  for (int j = 0; j < HH; ++j) {
    float v = hbr[j] * 100.f;
    sa += v * Wa[HH + j];
    sb += v * Wb[HH + j];
  }
  sa += ba[0];
  sb += bb[0];
  float a  = fminf(fmaxf(softplusf_(sa), 1e-6f), 100.f);
  float bk = fminf(fmaxf(softplusf_(sb), 1e-6f), 100.f);
  float opa = 1.f + 1.f / a;
  float lb  = lgammaf(opa) + lgammaf(bk) - lgammaf(opa + bk);
  float mean = bk * expf(lb);
  mean = fminf(fmaxf(-0.1f + 1.2f * mean, 0.f), 1.f);
  zp[idx] = mask[idx] ? mean : 0.f;
}

// ---------------------------------------------------------------------------
// Kernel 5: stable top-k per batch row (rank = stable descending argsort pos)
// ---------------------------------------------------------------------------
__global__ __launch_bounds__(256)
void topk_kernel(const float* __restrict__ zp,
                 const unsigned char* __restrict__ mask,
                 float* __restrict__ out) {
  __shared__ float z[TT];
  __shared__ int kk;
  int b = blockIdx.x;
  for (int t = threadIdx.x; t < TT; t += blockDim.x) z[t] = zp[b * TT + t];
  if (threadIdx.x == 0) {
    int len = 0;
    for (int t = 0; t < TT; ++t) len += mask[b * TT + t] ? 1 : 0;
    kk = (int)rintf(0.1f * (float)len);   // BUDGET/100 * length, RNE = jnp.round
  }
  __syncthreads();
  for (int t = threadIdx.x; t < TT; t += blockDim.x) {
    float zt = z[t];
    int rank = 0;
    for (int u = 0; u < TT; ++u) {
      float zu = z[u];
      rank += (zu > zt || (zu == zt && u < t)) ? 1 : 0;
    }
    bool sel = (rank < kk) && (zt > 0.f);
    out[b * TT + t] = mask[b * TT + t] ? (sel ? 1.f : 0.f) : 0.f;
  }
}

// ---------------------------------------------------------------------------
// Host launcher
// ---------------------------------------------------------------------------
extern "C" void kernel_launch(void* const* d_in, const int* in_sizes, int n_in,
                              void* d_out, int out_size, void* d_ws, size_t ws_size,
                              hipStream_t stream) {
  const int*           x     = (const int*)d_in[0];
  const unsigned char* mask  = (const unsigned char*)d_in[1];   // numpy bool
  const float*         embed = (const float*)d_in[2];
  const float*         Wih_f = (const float*)d_in[3];
  const float*         Whh_f = (const float*)d_in[4];
  const float*         b_f   = (const float*)d_in[5];
  const float*         Wih_b = (const float*)d_in[6];
  const float*         Whh_b = (const float*)d_in[7];
  const float*         b_b   = (const float*)d_in[8];
  const float*         Wa    = (const float*)d_in[9];
  const float*         ba    = (const float*)d_in[10];
  const float*         Wb    = (const float*)d_in[11];
  const float*         bb    = (const float*)d_in[12];
  float* out = (float*)d_out;

  // workspace carve-out (256-B aligned)
  char* ws = (char*)d_ws;
  size_t off = 0;
  auto carve = [&](size_t bytes) -> void* {
    off = (off + 255) & ~(size_t)255;
    void* p = ws + off;
    off += bytes;
    return p;
  };
  unsigned short* emb_pad  = (unsigned short*)carve((size_t)MROWS * EP * 2);
  unsigned short* wih_f_p  = (unsigned short*)carve((size_t)NG * EP * 2);
  unsigned short* wih_b_p  = (unsigned short*)carve((size_t)NG * EP * 2);
  unsigned short* whh_f_p  = (unsigned short*)carve((size_t)NG * KPH * 2);
  unsigned short* whh_b_p  = (unsigned short*)carve((size_t)NG * KPH * 2);
  float*          gin_f    = (float*)carve((size_t)MROWS * NG * 4);
  float*          gin_b    = (float*)carve((size_t)MROWS * NG * 4);
  float*          hf       = (float*)carve((size_t)MROWS * HH * 4);
  float*          hb       = (float*)carve((size_t)MROWS * HH * 4);
  float*          zp       = (float*)carve((size_t)MROWS * 4);
  (void)ws_size; (void)in_sizes; (void)n_in; (void)out_size;

  // 1) staging: gather + bf16 pad
  {
    long long total = (long long)MROWS * EP;
    int blocks = (int)((total + 255) / 256);
    pad_embed_kernel<<<blocks, 256, 0, stream>>>(x, embed, emb_pad);
  }
  {
    long long total = (long long)NG * EP;
    int blocks = (int)((total + 255) / 256);
    pad_weight_kernel<<<blocks, 256, 0, stream>>>(Wih_f, wih_f_p, NG, EE, EP);
    pad_weight_kernel<<<blocks, 256, 0, stream>>>(Wih_b, wih_b_p, NG, EE, EP);
  }
  {
    long long total = (long long)NG * KPH;
    int blocks = (int)((total + 255) / 256);
    pad_weight_kernel<<<blocks, 256, 0, stream>>>(Whh_f, whh_f_p, NG, HH, KPH);
    pad_weight_kernel<<<blocks, 256, 0, stream>>>(Whh_b, whh_b_p, NG, HH, KPH);
  }

  // 2) big input-projection GEMMs (WMMA bf16 -> f32, async A staging)
  {
    dim3 grid((NG / 16 + 7) / 8, MROWS / 16);   // (7, 4096)
    input_proj_kernel<<<grid, 256, 0, stream>>>(emb_pad, wih_f_p, b_f, gin_f);
    input_proj_kernel<<<grid, 256, 0, stream>>>(emb_pad, wih_b_p, b_b, gin_b);
  }

  // 3) sequential bi-LSTM (one persistent workgroup per direction, TDM feed)
  lstm_kernel<<<2, 512, 0, stream>>>(whh_f_p, whh_b_p, gin_f, gin_b,
                                     mask, hf, hb);

  // 4) Kumaraswamy head
  head_kernel<<<(MROWS + 255) / 256, 256, 0, stream>>>(hf, hb, Wa, ba, Wb, bb,
                                                       mask, zp);

  // 5) per-row stable top-k selection
  topk_kernel<<<BB, 256, 0, stream>>>(zp, mask, out);
}
